// SNN_19559281066292
// MI455X (gfx1250) — matrix operand order repaired and checked by reading, not compiled, not run on hardware
//
#include <hip/hip_runtime.h>

// ---------------------------------------------------------------------------
// SNN forward on gfx1250: bf16 WMMA GEMMs + TDM-staged B operands in LDS
// + fused LIF dynamics.
// ---------------------------------------------------------------------------

typedef __attribute__((ext_vector_type(16))) __bf16 v16bf;
typedef __attribute__((ext_vector_type(8)))  __bf16 v8bf;
typedef __attribute__((ext_vector_type(8)))  float  v8f;
typedef __attribute__((ext_vector_type(4)))  unsigned int v4u;
typedef __attribute__((ext_vector_type(8)))  int    v8i;
typedef __attribute__((ext_vector_type(4)))  int    v4i;

#define ALPHA_C 0.90483741803595957f   // exp(-0.001/0.01)
#define KAPPA_C 0.90483741803595957f

static constexpr int B_   = 256;
static constexpr int T_   = 100;
static constexpr int DIN  = 1024;
static constexpr int H_   = 2048;
static constexpr int DOUT = 128;

static constexpr int FRAG     = 512;               // bf16 elements per 16x32 / 32x16 fragment
static constexpr int CHUNK_KT = 16;                // k-tiles staged per LDS chunk
static constexpr int CHUNK_EL = CHUNK_KT * FRAG;   // 8192 bf16 = 16 KB per column strip

#if __has_builtin(__builtin_amdgcn_tensor_load_to_lds)
#define USE_TDM 1
#else
#define USE_TDM 0
#endif

// ---------------------------------------------------------------------------
// Fragment packing (see ISA 7.12.2):
// A-fragment (16x32): lane = (m&15) | (((k>>3)&1)<<4);
//                     klocal = (e&7) + ((e&8)?16:0) + lane_hi*8
// B-fragment (32x16): lane = (n&15) | (((k>>3)&1)<<4);
//                     klocal = (e&7) + ((e&8)?16:0) + lane_hi*8
// Each fragment = 512 bf16 (1 KB); each lane's 16 bf16 are contiguous.
// ---------------------------------------------------------------------------

__global__ void pack_B_kernel(const float* __restrict__ W, __bf16* __restrict__ out,
                              int K, int N) {
    int p = blockIdx.x * 256 + threadIdx.x;
    int total = K * N;
    if (p >= total) return;
    int frag = p >> 9;
    int w    = p & 511;
    int lane = w >> 4;
    int e    = w & 15;
    int Ktiles = K >> 5;
    int nt = frag / Ktiles;
    int kt = frag - nt * Ktiles;
    int n  = nt * 16 + (lane & 15);
    int k  = kt * 32 + (e & 7) + ((e & 8) ? 16 : 0) + ((lane >> 4) << 3);
    out[p] = (__bf16)W[(size_t)k * N + n];
}

__global__ void pack_x_kernel(const float* __restrict__ x, __bf16* __restrict__ out,
                              int t) {
    int p = blockIdx.x * 256 + threadIdx.x;   // total = B_*DIN
    int frag = p >> 9;
    int w    = p & 511;
    int lane = w >> 4;
    int e    = w & 15;
    const int Ktiles = DIN >> 5;              // 32
    int mt16 = frag / Ktiles;
    int kt   = frag - mt16 * Ktiles;
    int m    = mt16 * 16 + (lane & 15);
    int k    = kt * 32 + (e & 7) + ((e & 8) ? 16 : 0) + ((lane >> 4) << 3);
    out[p] = (__bf16)x[((size_t)m * T_ + t) * DIN + k];
}

__device__ __forceinline__ void store_zpack(__bf16* __restrict__ Zpack,
                                            int m, int k, float z) {
    const int Ktiles = H_ >> 5;               // 64
    int mt16 = m >> 4;
    int kt   = k >> 5;
    int lane = (m & 15) | (((k >> 3) & 1) << 4);
    int kl   = k & 31;
    int e    = (kl & 7) | ((kl & 16) ? 8 : 0);
    Zpack[((size_t)(mt16 * Ktiles + kt)) * 512 + lane * 16 + e] = (__bf16)z;
}

// ---------------------------------------------------------------------------
// Tensor Data Mover: 1-D contiguous copy global -> LDS.
// D# group0: count=1, lds_addr, 57-bit global addr, type=2 ("image").
// D# group1: data_size=3 (8B units), tensor_dim0 = tile_dim0 = nUnits,
//            tensor_dim1 = 1, tensor_dim0_stride = nUnits. Groups 2/3 zero.
// ---------------------------------------------------------------------------
__device__ __forceinline__ unsigned lds_offset_of(const void* p) {
    return (unsigned)(uintptr_t)p;   // low 32 bits of generic LDS pointer = LDS byte addr
}

#if USE_TDM
__device__ __forceinline__ void tdm_load_1d(unsigned ldsOff, const void* gptr,
                                            unsigned nUnits8) {
    unsigned long long ga = (unsigned long long)(uintptr_t)gptr;
    v4u g0;
    g0[0] = 1u;                                        // count = 1 valid descriptor
    g0[1] = ldsOff;                                    // lds_addr
    g0[2] = (unsigned)(ga & 0xFFFFFFFFu);              // global_addr[31:0]
    g0[3] = (unsigned)((ga >> 32) & 0x1FFFFFFu) | (2u << 30);  // addr[56:32] | type=2
    v8i g1;
    g1[0] = (int)(3u << 16);                           // data_size = 8 bytes
    g1[1] = (int)((nUnits8 & 0xFFFFu) << 16);          // tensor_dim0[15:0]
    g1[2] = (int)(((nUnits8 >> 16) & 0xFFFFu) | (1u << 16)); // td0[31:16] | td1 lo = 1
    g1[3] = (int)((nUnits8 & 0xFFFFu) << 16);          // tile_dim0 (td1 hi = 0)
    g1[4] = 0;                                         // tile_dim1 = tile_dim2 = 0
    g1[5] = (int)nUnits8;                              // tensor_dim0_stride[31:0]
    g1[6] = 0;
    g1[7] = 0;
    v4i z4 = {0, 0, 0, 0};
    v8i z8 = {0, 0, 0, 0, 0, 0, 0, 0};
    // 6-arg toolchain variant (clang-23 / therock-10.0 headers).
    __builtin_amdgcn_tensor_load_to_lds(g0, g1, z4, z4, z8, 0);
}
#endif

// Issue staging of one chunk (two 16 KB column strips) into LDS buffer `dst`.
__device__ __forceinline__ void stage_issue(__bf16* dst,
                                            const __bf16* b0s, const __bf16* b1s,
                                            int ch, int waveid, int tid) {
#if USE_TDM
    if (waveid == 0) {
        tdm_load_1d(lds_offset_of(dst),
                    b0s + (size_t)ch * CHUNK_EL, (CHUNK_EL * 2) / 8);
        tdm_load_1d(lds_offset_of(dst + CHUNK_EL),
                    b1s + (size_t)ch * CHUNK_EL, (CHUNK_EL * 2) / 8);
    }
#else
    const v8bf* s0 = (const v8bf*)(b0s + (size_t)ch * CHUNK_EL);
    const v8bf* s1 = (const v8bf*)(b1s + (size_t)ch * CHUNK_EL);
    v8bf* d0 = (v8bf*)dst;
    v8bf* d1 = (v8bf*)(dst + CHUNK_EL);
#pragma unroll
    for (int i = 0; i < CHUNK_EL / 8 / 256; ++i) {   // 4 x 16B per thread per strip
        d0[tid + i * 256] = s0[tid + i * 256];
        d1[tid + i * 256] = s1[tid + i * 256];
    }
#endif
}

__device__ __forceinline__ void stage_wait(int waveid) {
#if USE_TDM
    if (waveid == 0) __builtin_amdgcn_s_wait_tensorcnt(0);
#endif
}

// ---------------------------------------------------------------------------
// One wave accumulates a 32x32 f32 tile.  B fragments come from LDS (staged
// per block, double-buffered); A fragments stream from global (per-wave).
// sB points to __shared__ [2][2*CHUNK_EL].
// ---------------------------------------------------------------------------
__device__ __forceinline__ void gemm_lds_32x32(v8f (&c)[2][2],
                                               const __bf16* __restrict__ Apack,
                                               const __bf16* __restrict__ Bpack,
                                               int Ktiles, int mt, int nt,
                                               int lane, int waveid, int tid,
                                               __bf16* sB) {
    const __bf16* b0s = Bpack + (size_t)(2 * nt)     * Ktiles * FRAG;
    const __bf16* b1s = Bpack + (size_t)(2 * nt + 1) * Ktiles * FRAG;
    const __bf16* a0p = Apack + (size_t)(2 * mt)     * Ktiles * FRAG + lane * 16;
    const __bf16* a1p = Apack + (size_t)(2 * mt + 1) * Ktiles * FRAG + lane * 16;
    const int nchunks = Ktiles / CHUNK_KT;

    stage_issue(sB, b0s, b1s, 0, waveid, tid);
    stage_wait(waveid);
    __syncthreads();

    for (int ch = 0; ch < nchunks; ++ch) {
        __bf16* cur = sB + (size_t)(ch & 1) * (2 * CHUNK_EL);
        if (ch + 1 < nchunks)
            stage_issue(sB + (size_t)((ch + 1) & 1) * (2 * CHUNK_EL),
                        b0s, b1s, ch + 1, waveid, tid);
#pragma unroll 2
        for (int kk = 0; kk < CHUNK_KT; ++kk) {
            int kt = ch * CHUNK_KT + kk;
            v16bf a0 = *(const v16bf*)(a0p + (size_t)kt * FRAG);
            v16bf a1 = *(const v16bf*)(a1p + (size_t)kt * FRAG);
            v16bf b0 = *(const v16bf*)(cur + kk * FRAG + lane * 16);
            v16bf b1 = *(const v16bf*)(cur + CHUNK_EL + kk * FRAG + lane * 16);
            if (kt + CHUNK_KT < Ktiles) {   // warm next chunk's A into WGP cache
                __builtin_prefetch(a0p + (size_t)(kt + CHUNK_KT) * FRAG, 0, 3);
                __builtin_prefetch(a1p + (size_t)(kt + CHUNK_KT) * FRAG, 0, 3);
            }
            c[0][0] = __builtin_amdgcn_wmma_f32_16x16x32_bf16(false, a0, false, b0,
                                                              (short)0, c[0][0], false, false);
            c[0][1] = __builtin_amdgcn_wmma_f32_16x16x32_bf16(false, a0, false, b1,
                                                              (short)0, c[0][1], false, false);
            c[1][0] = __builtin_amdgcn_wmma_f32_16x16x32_bf16(false, a1, false, b0,
                                                              (short)0, c[1][0], false, false);
            c[1][1] = __builtin_amdgcn_wmma_f32_16x16x32_bf16(false, a1, false, b1,
                                                              (short)0, c[1][1], false, false);
        }
        stage_wait(waveid);
        __syncthreads();
    }
}

// ---------------------------------------------------------------------------
// LIF layer step: I = Ain@Bin + (t>0 ? Arec@Brec : 0)
//                 V' = ALPHA*Vprev*(1-Zprev) + I ; Z' = (V' > 1)
// Grid: 64 blocks (one 32-col strip each) x 256 threads (8 waves = 8 M-tiles).
// ---------------------------------------------------------------------------
__global__ void __launch_bounds__(256)
lif_layer_kernel(const __bf16* __restrict__ ApackIn,
                 const __bf16* __restrict__ BpackIn, int KtilesIn,
                 const __bf16* __restrict__ ApackRec,
                 const __bf16* __restrict__ BpackRec,
                 float* __restrict__ Vsec, float* __restrict__ Zsec,
                 __bf16* __restrict__ ZpackOut, int t) {
    __shared__ __align__(1024) __bf16 sB[2][2 * CHUNK_EL];   // 64 KB
    int tid    = threadIdx.x;
    int lane   = tid & 31;
    int waveid = tid >> 5;
    int mt     = waveid;        // M = 256 -> 8 tiles of 32
    int nt     = blockIdx.x;    // N = 2048 -> 64 strips of 32

    v8f zero = {0.f, 0.f, 0.f, 0.f, 0.f, 0.f, 0.f, 0.f};
    v8f c[2][2] = {{zero, zero}, {zero, zero}};

    gemm_lds_32x32(c, ApackIn, BpackIn, KtilesIn, mt, nt, lane, waveid, tid, &sB[0][0]);
    if (t > 0)   // uniform per block -> barrier-safe
        gemm_lds_32x32(c, ApackRec, BpackRec, H_ >> 5, mt, nt, lane, waveid, tid, &sB[0][0]);

    const int mbase = mt * 32;
    const int nbase = nt * 32;
    const size_t rowStride = (size_t)T_ * H_;
    const size_t tOff  = (size_t)t * H_;
    const size_t tPrev = (size_t)(t - 1) * H_;

#pragma unroll
    for (int tm = 0; tm < 2; ++tm)
#pragma unroll
    for (int tn = 0; tn < 2; ++tn)
#pragma unroll
    for (int r = 0; r < 8; ++r) {
        int m = mbase + tm * 16 + r + ((lane >> 4) << 3);
        int n = nbase + tn * 16 + (lane & 15);
        float I = c[tm][tn][r];
        float Vn;
        if (t > 0) {
            size_t pidx = (size_t)m * rowStride + tPrev + n;
            float Vp = Vsec[pidx];
            float Zp = Zsec[pidx];
            Vn = ALPHA_C * Vp * (1.0f - Zp) + I;
        } else {
            Vn = I;
        }
        float Zn = (Vn > 1.0f) ? 1.0f : 0.0f;
        size_t idx = (size_t)m * rowStride + tOff + n;
        Vsec[idx] = Vn;
        Zsec[idx] = Zn;
        store_zpack(ZpackOut, m, n, Zn);
    }
}

// ---------------------------------------------------------------------------
// Readout: Vo' = KAPPA*Vo + Z2@W_out + b_out.  Grid: 4 blocks x 256 threads.
// ---------------------------------------------------------------------------
__global__ void __launch_bounds__(256)
readout_kernel(const __bf16* __restrict__ Apack,
               const __bf16* __restrict__ Bpack,
               const float* __restrict__ b_out,
               float* __restrict__ Osec, int t) {
    __shared__ __align__(1024) __bf16 sB[2][2 * CHUNK_EL];
    int tid    = threadIdx.x;
    int lane   = tid & 31;
    int waveid = tid >> 5;
    int mt     = waveid;        // M = 256 -> 8 tiles
    int nt     = blockIdx.x;    // N = 128 -> 4 strips

    v8f zero = {0.f, 0.f, 0.f, 0.f, 0.f, 0.f, 0.f, 0.f};
    v8f c[2][2] = {{zero, zero}, {zero, zero}};

    gemm_lds_32x32(c, Apack, Bpack, H_ >> 5, mt, nt, lane, waveid, tid, &sB[0][0]);

    const int mbase = mt * 32;
    const int nbase = nt * 32;
    const size_t rowStride = (size_t)T_ * DOUT;

#pragma unroll
    for (int tm = 0; tm < 2; ++tm)
#pragma unroll
    for (int tn = 0; tn < 2; ++tn)
#pragma unroll
    for (int r = 0; r < 8; ++r) {
        int m = mbase + tm * 16 + r + ((lane >> 4) << 3);
        int n = nbase + tn * 16 + (lane & 15);
        float acc = c[tm][tn][r] + b_out[n];
        if (t > 0)
            acc += KAPPA_C * Osec[(size_t)m * rowStride + (size_t)(t - 1) * DOUT + n];
        Osec[(size_t)m * rowStride + (size_t)t * DOUT + n] = acc;
    }
}

// ---------------------------------------------------------------------------
// Host orchestration: pack weights once, then 100 sequential LIF steps.
// Workspace use: ~33 MB of bf16 packed buffers.
// ---------------------------------------------------------------------------
extern "C" void kernel_launch(void* const* d_in, const int* in_sizes, int n_in,
                              void* d_out, int out_size, void* d_ws, size_t ws_size,
                              hipStream_t stream) {
    const float* inputs = (const float*)d_in[0];
    const float* W_in   = (const float*)d_in[1];
    const float* R1     = (const float*)d_in[2];
    const float* W_h    = (const float*)d_in[3];
    const float* R2     = (const float*)d_in[4];
    const float* W_out  = (const float*)d_in[5];
    const float* b_out  = (const float*)d_in[6];

    // d_out = concat(out[B,T,DOUT], V1[B,T,H], Z1, V2, Z2), all f32.
    float* base  = (float*)d_out;
    float* OUTs  = base;
    float* V1s   = OUTs + (size_t)B_ * T_ * DOUT;
    float* Z1s   = V1s  + (size_t)B_ * T_ * H_;
    float* V2s   = Z1s  + (size_t)B_ * T_ * H_;
    float* Z2s   = V2s  + (size_t)B_ * T_ * H_;

    // Workspace layout (bf16 elements).
    __bf16* ws   = (__bf16*)d_ws;
    __bf16* WinP = ws;                               // 1024*2048
    __bf16* R1P  = WinP + (size_t)DIN * H_;          // 2048*2048
    __bf16* WhP  = R1P  + (size_t)H_ * H_;
    __bf16* R2P  = WhP  + (size_t)H_ * H_;
    __bf16* WoP  = R2P  + (size_t)H_ * H_;           // 2048*128
    __bf16* xP   = WoP  + (size_t)H_ * DOUT;         // 256*1024
    __bf16* Z1P0 = xP   + (size_t)B_ * DIN;          // 256*2048 each
    __bf16* Z1P1 = Z1P0 + (size_t)B_ * H_;
    __bf16* Z2P0 = Z1P1 + (size_t)B_ * H_;
    __bf16* Z2P1 = Z2P0 + (size_t)B_ * H_;
    __bf16* Z1P[2] = {Z1P0, Z1P1};
    __bf16* Z2P[2] = {Z2P0, Z2P1};

    // One-time weight packing (f32 -> bf16 fragment order).
    pack_B_kernel<<<(DIN * H_) / 256, 256, 0, stream>>>(W_in, WinP, DIN, H_);
    pack_B_kernel<<<(H_ * H_) / 256, 256, 0, stream>>>(R1, R1P, H_, H_);
    pack_B_kernel<<<(H_ * H_) / 256, 256, 0, stream>>>(W_h, WhP, H_, H_);
    pack_B_kernel<<<(H_ * H_) / 256, 256, 0, stream>>>(R2, R2P, H_, H_);
    pack_B_kernel<<<(H_ * DOUT) / 256, 256, 0, stream>>>(W_out, WoP, H_, DOUT);

    for (int t = 0; t < T_; ++t) {
        int cur = t & 1, prev = cur ^ 1;
        pack_x_kernel<<<(B_ * DIN) / 256, 256, 0, stream>>>(inputs, xP, t);
        // Layer 1: x_t@W_in + Z1(t-1)@R1
        lif_layer_kernel<<<64, 256, 0, stream>>>(xP, WinP, DIN >> 5,
                                                 Z1P[prev], R1P,
                                                 V1s, Z1s, Z1P[cur], t);
        // Layer 2: Z1(t)@W_h + Z2(t-1)@R2
        lif_layer_kernel<<<64, 256, 0, stream>>>(Z1P[cur], WhP, H_ >> 5,
                                                 Z2P[prev], R2P,
                                                 V2s, Z2s, Z2P[cur], t);
        // Readout: KAPPA*Vo + Z2(t)@W_out + b
        readout_kernel<<<4, 256, 0, stream>>>(Z2P[cur], WoP, b_out, OUTs, t);
    }
}
